// DIEN_41360535060900
// MI455X (gfx1250) — matrix operand order repaired and checked by reading, not compiled, not run on hardware
//
#include <hip/hip_runtime.h>
#include <hip/hip_bf16.h>
#include <stdint.h>

#define B_ 4096
#define T_ 50
#define D_ 128
#define U_ 128

#if defined(__HIP_DEVICE_COMPILE__) && \
    __has_builtin(__builtin_amdgcn_global_load_async_to_lds_b128) && \
    __has_builtin(__builtin_amdgcn_s_wait_asynccnt)
#define HAVE_ASYNC_LDS 1
#else
#define HAVE_ASYNC_LDS 0
#endif

typedef __attribute__((ext_vector_type(16))) __bf16 v16bf;
typedef __attribute__((ext_vector_type(8)))  float  v8f;
typedef int v4i __attribute__((vector_size(16)));

union FragAB { v16bf v; unsigned int u[8]; uint4 q[2]; };
union FragC  { v8f v; float f[8]; };

__device__ __forceinline__ unsigned short f2bf(float x){
  unsigned int u = __float_as_uint(x);
  return (unsigned short)((u + 0x7FFFu + ((u >> 16) & 1u)) >> 16);
}
__device__ __forceinline__ float bf2f(unsigned short h){
  return __uint_as_float(((unsigned int)h) << 16);
}
__device__ __forceinline__ float sigm(float x){ return 1.0f / (1.0f + expf(-x)); }

#if HAVE_ASYNC_LDS
__device__ __forceinline__ void async_copy16(const void* gsrc, void* ldst){
  __builtin_amdgcn_global_load_async_to_lds_b128(
      (__attribute__((address_space(1))) v4i*)gsrc,
      (__attribute__((address_space(3))) v4i*)ldst, 0, 0);
}
__device__ __forceinline__ void async_wait(){ __builtin_amdgcn_s_wait_asynccnt(0); }
#endif

// A fragment (16x32 bf16): lane holds row M=lane%16; element e -> K = (e%8) + 8*(lane/16) + 16*(e/8)
// consecutive element pairs are consecutive K -> gather as 8 x b32 (pairs coalesce to 2 x b128).
__device__ __forceinline__ v16bf load_a_frag(const unsigned short* rowbase, int ldk, int kt){
  const int lane = threadIdx.x & 31;
  const int row  = lane & 15;
  const int half = lane >> 4;
  FragAB f;
  const unsigned int* p = (const unsigned int*)(rowbase + row * ldk + kt * 32);
#pragma unroll
  for (int q = 0; q < 8; ++q){
    int idx = q + 4 * half + ((q >= 4) ? 4 : 0);   // pair index = k/2
    f.u[q] = p[idx];
  }
  return f.v;
}

// B fragment (32x16 bf16) pre-packed contiguously: lane reads 16 bf16 = 2 x b128.
__device__ __forceinline__ v16bf load_b_frag(const unsigned short* packed, int frag){
  const int lane = threadIdx.x & 31;
  FragAB f;
  const uint4* p = (const uint4*)(packed + (size_t)frag * 512 + lane * 16);
  f.q[0] = p[0]; f.q[1] = p[1];
  return f.v;
}

__device__ __forceinline__ v8f wmma_bf16(v16bf a, v16bf b, v8f c){
  return __builtin_amdgcn_wmma_f32_16x16x32_bf16(false, a, false, b, (short)0, c, false, false);
}

// ---------------------------------------------------------------------------
// Pack an f32 weight sub-matrix into bf16 B-fragment layout.
// dest fragment (ktd=k_off+kti, ntd=n_off+nti) at frag index ntd*KT_dst+ktd.
// B layout: lane L: n=L%16, half=L/16, element e -> K = (e%8) + 8*half + 16*(e/8).
// ---------------------------------------------------------------------------
__global__ void pack_b_kernel(unsigned short* __restrict__ dst, int KT_dst,
                              const float* __restrict__ src, int Nsrc,
                              int k_off, int n_off, int kt_cnt, int nt_cnt, int src_n_off)
{
  int fid = blockIdx.x * (blockDim.x >> 5) + (threadIdx.x >> 5);
  if (fid >= kt_cnt * nt_cnt) return;
  int kti = fid % kt_cnt, nti = fid / kt_cnt;
  int lane = threadIdx.x & 31, half = lane >> 4, n = lane & 15;
  unsigned short* out = dst + ((size_t)(n_off + nti) * KT_dst + (k_off + kti)) * 512 + lane * 16;
#pragma unroll
  for (int e = 0; e < 16; ++e){
    int k = kti * 32 + (e & 7) + 8 * half + 16 * (e >> 3);
    out[e] = f2bf(src[(size_t)k * Nsrc + src_n_off + nti * 16 + n]);
  }
}

__global__ void bnprep_kernel(const float* __restrict__ g, const float* __restrict__ be,
                              const float* __restrict__ mu, const float* __restrict__ va,
                              float* __restrict__ ss)
{
  int i = threadIdx.x;
  if (i < 256){
    float s = g[i] * rsqrtf(va[i] + 1.0e-3f);
    ss[i] = s;
    ss[256 + i] = be[i] - mu[i] * s;
  }
}

// ---------------------------------------------------------------------------
// Recurrent scan. MODE 0 = GRU (x from inputs, writes hs bf16),
//                 MODE 1 = AUGRU (x from hs bf16, reads ats, writes h_final).
// Per step: pre = [h | x](32x256) @ packed weights -> 512 preact columns:
//   [0:128)   z/u gate (h-part + x-part summed)
//   [128:256) r gate   (summed)
//   [256:384) x @ Wh   (xh / xc)
//   [384:512) h @ Uh   (hh / h@Uc)
// Packed: frags 0..127 = Wzr (KT=8, 16 nt), 128..159 = Wh (KT=4, 8 nt),
//         160..191 = Uh (KT=4, 8 nt).
// Wave w: msub = w>>2 (16 rows), q = w&3 selects an nt group with uniform
// kt-range so preloaded A fragments are indexed by constants.
// ---------------------------------------------------------------------------
template<int MODE>
__global__ void __launch_bounds__(256, 1) scan_kernel(
    const float* __restrict__ inputs, unsigned short* __restrict__ hs,
    const unsigned short* __restrict__ wpack,
    const float* __restrict__ b0, const float* __restrict__ b1, const float* __restrict__ b2,
    const float* __restrict__ ats, float* __restrict__ hfinal)
{
  __shared__ __align__(16) unsigned short sW[192 * 512];   // 192 KB packed weights
  __shared__ __align__(16) float          sPre[32 * 512];  // 64 KB preactivations
  __shared__ __align__(16) float          sH[32 * 128];    // 16 KB f32 hidden state
  __shared__ __align__(16) unsigned short sA[32 * 256];    // 16 KB bf16 [h | x]

  const int tid   = threadIdx.x;
  const int brow0 = blockIdx.x * 32;

  { // stage weights into LDS once (async copy when available)
#if HAVE_ASYNC_LDS
    for (int i = tid; i < (192 * 512) / 8; i += 256)
      async_copy16((const void*)((const uint4*)wpack + i), (void*)((uint4*)sW + i));
#else
    const uint4* s = (const uint4*)wpack; uint4* d = (uint4*)sW;
    for (int i = tid; i < (192 * 512) / 8; i += 256) d[i] = s[i];
#endif
  }
  for (int i = tid; i < 32 * 128; i += 256){
    sH[i] = 0.0f;
    int r = i >> 7, u = i & 127;
    sA[r * 256 + u] = 0;
  }
#if HAVE_ASYNC_LDS
  async_wait();
#endif
  __syncthreads();

  const int wave = tid >> 5, lane = tid & 31, half = lane >> 4, ncol = lane & 15;
  const int q = wave & 3, msub = wave >> 2;
  const int srow = tid >> 3, sseg = tid & 7;   // staging: 8 threads per row

  for (int t = 0; t < T_; ++t){
    // ---- stage x_t (cols 128..255 of A), 32 rows ----
    if (MODE == 0){
      const float* src = inputs + ((size_t)(brow0 + srow) * (T_ + 1) + t) * D_ + sseg * 16;
      unsigned short* dst = sA + srow * 256 + 128 + sseg * 16;
#pragma unroll
      for (int d = 0; d < 16; ++d) dst[d] = f2bf(src[d]);
      if (t + 1 < T_)
        __builtin_prefetch(inputs + ((size_t)(brow0 + srow) * (T_ + 1) + t + 1) * D_ + sseg * 16, 0, 3);
    } else {
      const unsigned short* src = hs + ((size_t)(brow0 + srow) * T_ + t) * U_ + sseg * 16;
      unsigned short* dst = sA + srow * 256 + 128 + sseg * 16;
#if HAVE_ASYNC_LDS
      async_copy16((const void*)src, (void*)dst);
      async_copy16((const void*)(src + 8), (void*)(dst + 8));
      async_wait();
#else
#pragma unroll
      for (int d = 0; d < 16; ++d) dst[d] = src[d];
#endif
      if (t + 1 < T_)
        __builtin_prefetch(hs + ((size_t)(brow0 + srow) * T_ + t + 1) * U_ + sseg * 16, 0, 3);
    }
    __syncthreads();

    // ---- GEMM: preload this wave's 8 A fragments, then its 8 n-tiles ----
    {
      v16bf afr[8];
#pragma unroll
      for (int kt = 0; kt < 8; ++kt) afr[kt] = load_a_frag(sA + msub * 16 * 256, 256, kt);

      if (q < 2){                       // nt = q*8 .. q*8+7 : Wzr, K = 256
#pragma unroll
        for (int i = 0; i < 8; ++i){
          int nt = q * 8 + i;
          FragC c;
#pragma unroll
          for (int j = 0; j < 8; ++j) c.f[j] = 0.0f;
#pragma unroll
          for (int kt = 0; kt < 8; ++kt) c.v = wmma_bf16(afr[kt], load_b_frag(sW, nt * 8 + kt), c.v);
#pragma unroll
          for (int j = 0; j < 8; ++j)
            sPre[(msub * 16 + j + 8 * half) * 512 + nt * 16 + ncol] = c.f[j];
        }
      } else if (q == 2){               // nt = 16..23 : x @ Wh, A kt 4..7
#pragma unroll
        for (int i = 0; i < 8; ++i){
          int nt = 16 + i;
          FragC c;
#pragma unroll
          for (int j = 0; j < 8; ++j) c.f[j] = 0.0f;
#pragma unroll
          for (int kt = 0; kt < 4; ++kt) c.v = wmma_bf16(afr[4 + kt], load_b_frag(sW, 128 + i * 4 + kt), c.v);
#pragma unroll
          for (int j = 0; j < 8; ++j)
            sPre[(msub * 16 + j + 8 * half) * 512 + nt * 16 + ncol] = c.f[j];
        }
      } else {                          // nt = 24..31 : h @ Uh, A kt 0..3
#pragma unroll
        for (int i = 0; i < 8; ++i){
          int nt = 24 + i;
          FragC c;
#pragma unroll
          for (int j = 0; j < 8; ++j) c.f[j] = 0.0f;
#pragma unroll
          for (int kt = 0; kt < 4; ++kt) c.v = wmma_bf16(afr[kt], load_b_frag(sW, 160 + i * 4 + kt), c.v);
#pragma unroll
          for (int j = 0; j < 8; ++j)
            sPre[(msub * 16 + j + 8 * half) * 512 + nt * 16 + ncol] = c.f[j];
        }
      }
    }
    __syncthreads();

    // ---- elementwise gate update ----
    for (int i = tid; i < 32 * 128; i += 256){
      int r = i >> 7, u = i & 127;
      const float* pre = &sPre[r * 512];
      float hold = sH[i], hn;
      if (MODE == 0){
        float z  = sigm(pre[u]       + b0[u]       + b1[u]);
        float rg = sigm(pre[128 + u] + b0[128 + u] + b1[128 + u]);
        float hc = tanhf(pre[256 + u] + b0[256 + u] + rg * (pre[384 + u] + b1[256 + u]));
        hn = z * hold + (1.0f - z) * hc;
        hs[((size_t)(brow0 + r) * T_ + t) * U_ + u] = f2bf(hn);
      } else {
        float ug = sigm(pre[u]       + b0[u]);
        float rg = sigm(pre[128 + u] + b1[u]);
        float cg = tanhf(pre[256 + u] + b2[u] + rg * pre[384 + u]);
        float at = ats[(size_t)(brow0 + r) * T_ + t];
        float uu = at * ug;
        hn = (1.0f - uu) * hold + uu * cg;
      }
      sH[i] = hn;
      sA[r * 256 + u] = f2bf(hn);
    }
    __syncthreads();
  }
  if (MODE == 1){
    for (int i = tid; i < 32 * 128; i += 256){
      int r = i >> 7, u = i & 127;
      hfinal[(size_t)(brow0 + r) * U_ + u] = sH[i];
    }
  }
}

// ---------------------------------------------------------------------------
// Generic GEMM: C(M,N) = act(A(M,K) @ Wpacked + bias). 64 rows/block, 8 waves.
// AMODE 0: A = f32 (or bf16 if a_bf16) row-major, ld = K
// AMODE 2: A = emb built on the fly from hs(bf16) + news (K=512)
// AMODE 3: A = BN(concat(h_final, news)) via aux1 scale/shift (K=256)
// ---------------------------------------------------------------------------
template<int K, int AMODE>
__global__ void __launch_bounds__(256) gemm_kernel(
    const void* __restrict__ aptr, const unsigned short* __restrict__ wp,
    const float* __restrict__ bias, void* __restrict__ outp,
    const float* __restrict__ aux0, const float* __restrict__ aux1,
    int N, int act, int a_bf16, int out_bf16)
{
  __shared__ __align__(16) unsigned short sA[64 * K];
  const int tid = threadIdx.x;
  const int m0  = blockIdx.x * 64;

  { // stage A: 4 threads per row, contiguous chunks
    const int r  = tid >> 2;
    const int c0 = (tid & 3) * (K / 4);
    const int m  = m0 + r;
    if (AMODE == 0){
      if (a_bf16){
        const unsigned short* src = (const unsigned short*)aptr + (size_t)m * K + c0;
        unsigned short* dst = sA + r * K + c0;
#if HAVE_ASYNC_LDS
#pragma unroll
        for (int d = 0; d < K / 4; d += 8)
          async_copy16((const void*)(src + d), (void*)(dst + d));
        async_wait();
#else
#pragma unroll 8
        for (int d = 0; d < K / 4; ++d) dst[d] = src[d];
#endif
      } else {
        const float* src = (const float*)aptr + (size_t)m * K + c0;
#pragma unroll 8
        for (int d = 0; d < K / 4; ++d) sA[r * K + c0 + d] = f2bf(src[d]);
      }
    } else if (AMODE == 2){
      const int b = m / T_;
      const unsigned short* hrow = (const unsigned short*)aptr + (size_t)m * U_;
      const float* nrow = aux0 + ((size_t)b * (T_ + 1) + T_) * D_;
      const int sec = tid & 3;         // K/4 == 128 == one emb section per thread
#pragma unroll 8
      for (int dd = 0; dd < 128; ++dd){
        float hv = bf2f(hrow[dd]);
        float nv = nrow[dd];
        float v = (sec == 0) ? hv : (sec == 1) ? (nv - hv) : (sec == 2) ? nv : hv * nv;
        sA[r * K + c0 + dd] = f2bf(v);
      }
    } else { // AMODE == 3
#pragma unroll 8
      for (int dd = 0; dd < K / 4; ++dd){
        int d = c0 + dd;
        float x = (d < 128) ? ((const float*)aptr)[(size_t)m * U_ + d]
                            : aux0[((size_t)m * (T_ + 1) + T_) * D_ + (d - 128)];
        sA[r * K + d] = f2bf(x * aux1[d] + aux1[256 + d]);
      }
    }
  }
  __syncthreads();

  const int wave = tid >> 5, lane = tid & 31, half = lane >> 4, nn = lane & 15;
  const int msub = wave & 3, nhalf = wave >> 2;
  const int NT = N >> 4;
  constexpr int KT  = K / 32;
  constexpr int KTP = (KT <= 8) ? KT : 8;     // A fragments held in registers

  v16bf afr[KTP];
#pragma unroll
  for (int kt = 0; kt < KTP; ++kt) afr[kt] = load_a_frag(sA + msub * 16 * K, K, kt);

  for (int nt = nhalf * (NT / 2); nt < (nhalf + 1) * (NT / 2); ++nt){
    FragC c;
#pragma unroll
    for (int j = 0; j < 8; ++j) c.f[j] = 0.0f;
#pragma unroll
    for (int kt = 0; kt < KTP; ++kt)
      c.v = wmma_bf16(afr[kt], load_b_frag(wp, nt * KT + kt), c.v);
#pragma unroll
    for (int kt = KTP; kt < KT; ++kt){
      v16bf a = load_a_frag(sA + msub * 16 * K, K, kt);
      c.v = wmma_bf16(a, load_b_frag(wp, nt * KT + kt), c.v);
    }
    int n = nt * 16 + nn;
    float bv = bias[n];
#pragma unroll
    for (int j = 0; j < 8; ++j){
      int mrow = m0 + msub * 16 + j + 8 * half;
      float val = c.f[j] + bv;
      if (act == 1)      val = val > 0.0f ? val : 0.0f;
      else if (act == 2) val = val >= 0.0f ? val : 3.0e-4f * val;
      if (out_bf16) ((unsigned short*)outp)[(size_t)mrow * N + n] = f2bf(val);
      else          ((float*)outp)[(size_t)mrow * N + n] = val;
    }
  }
}

__global__ void rowdot_sigmoid_kernel(const float* __restrict__ A, int ldk,
                                      const float* __restrict__ w, const float* __restrict__ b,
                                      float* __restrict__ out, int M)
{
  int m = blockIdx.x * blockDim.x + threadIdx.x;
  if (m >= M) return;
  const float* row = A + (size_t)m * ldk;
  float s = b[0];
  for (int i = 0; i < ldk; ++i) s += row[i] * w[i];
  out[m] = sigm(s);
}

extern "C" void kernel_launch(void* const* d_in, const int* in_sizes, int n_in,
                              void* d_out, int out_size, void* d_ws, size_t ws_size,
                              hipStream_t stream)
{
  (void)in_sizes; (void)n_in; (void)out_size; (void)ws_size;
  const float* inputs  = (const float*)d_in[0];
  const float* gru_W   = (const float*)d_in[1];
  const float* gru_U   = (const float*)d_in[2];
  const float* gru_b   = (const float*)d_in[3];
  const float* att_W1  = (const float*)d_in[4];
  const float* att_b1  = (const float*)d_in[5];
  const float* att_W2  = (const float*)d_in[6];
  const float* att_b2  = (const float*)d_in[7];
  const float* att_W3  = (const float*)d_in[8];
  const float* att_b3  = (const float*)d_in[9];
  const float* au_Wu   = (const float*)d_in[10];
  const float* au_bu   = (const float*)d_in[11];
  const float* au_Uu   = (const float*)d_in[12];
  const float* au_Wr   = (const float*)d_in[13];
  const float* au_br   = (const float*)d_in[14];
  const float* au_Ur   = (const float*)d_in[15];
  const float* au_Wc   = (const float*)d_in[16];
  const float* au_bc   = (const float*)d_in[17];
  const float* au_Uc   = (const float*)d_in[18];
  const float* bn_gamma= (const float*)d_in[19];
  const float* bn_beta = (const float*)d_in[20];
  const float* bn_mean = (const float*)d_in[21];
  const float* bn_var  = (const float*)d_in[22];
  const float* d_W1    = (const float*)d_in[23];
  const float* d_b1    = (const float*)d_in[24];
  const float* d_W2    = (const float*)d_in[25];
  const float* d_b2    = (const float*)d_in[26];
  const float* f_W     = (const float*)d_in[27];
  const float* f_b     = (const float*)d_in[28];

  char* ws = (char*)d_ws;
  size_t off = 0;
  auto alloc = [&](size_t bytes) -> char* {
    char* p = ws + off; off = (off + bytes + 255) & ~(size_t)255; return p;
  };
  unsigned short* pGRU = (unsigned short*)alloc(192 * 512 * 2);
  unsigned short* pAU  = (unsigned short*)alloc(192 * 512 * 2);
  unsigned short* pA1  = (unsigned short*)alloc(16 * 8 * 512 * 2);
  unsigned short* pA2  = (unsigned short*)alloc(4 * 4 * 512 * 2);
  unsigned short* pD1  = (unsigned short*)alloc(8 * 16 * 512 * 2);
  unsigned short* pD2  = (unsigned short*)alloc(8 * 8 * 512 * 2);
  float*          bnss = (float*)alloc(512 * 4);
  unsigned short* hsb  = (unsigned short*)alloc((size_t)B_ * T_ * U_ * 2);
  unsigned short* a1   = (unsigned short*)alloc((size_t)B_ * T_ * 128 * 2);
  float*          a2   = (float*)alloc((size_t)B_ * T_ * 64 * 4);
  float*          ats  = (float*)alloc((size_t)B_ * T_ * 4);
  float*          hfin = (float*)alloc((size_t)B_ * U_ * 4);
  float*          x1   = (float*)alloc((size_t)B_ * 256 * 4);
  float*          x2   = (float*)alloc((size_t)B_ * 128 * 4);

  auto packs = [&](unsigned short* dst, int KT, const float* src, int Nsrc,
                   int ko, int no, int kc, int nc, int sno){
    int frags = kc * nc;
    pack_b_kernel<<<(frags + 3) / 4, 128, 0, stream>>>(dst, KT, src, Nsrc, ko, no, kc, nc, sno);
  };
  // GRU combined weights: Wzr = [[Uz|Ur],[Wz|Wr]] (256x256), Wh = gru_W[:,256:], Uh = gru_U[:,256:]
  packs(pGRU,             8, gru_U, 384, 0, 0, 4, 16, 0);
  packs(pGRU,             8, gru_W, 384, 4, 0, 4, 16, 0);
  packs(pGRU + 128 * 512, 4, gru_W, 384, 0, 0, 4, 8, 256);
  packs(pGRU + 160 * 512, 4, gru_U, 384, 0, 0, 4, 8, 256);
  // AUGRU combined weights: Wzr = [[Uu|Ur],[Wu|Wr]], Wh = au_Wc, Uh = au_Uc
  packs(pAU,              8, au_Uu, 128, 0, 0, 4, 8, 0);
  packs(pAU,              8, au_Ur, 128, 0, 8, 4, 8, 0);
  packs(pAU,              8, au_Wu, 128, 4, 0, 4, 8, 0);
  packs(pAU,              8, au_Wr, 128, 4, 8, 4, 8, 0);
  packs(pAU + 128 * 512,  4, au_Wc, 128, 0, 0, 4, 8, 0);
  packs(pAU + 160 * 512,  4, au_Uc, 128, 0, 0, 4, 8, 0);
  // MLP weights
  packs(pA1, 16, att_W1, 128, 0, 0, 16, 8, 0);
  packs(pA2,  4, att_W2,  64, 0, 0, 4, 4, 0);
  packs(pD1,  8, d_W1,   256, 0, 0, 8, 16, 0);
  packs(pD2,  8, d_W2,   128, 0, 0, 8, 8, 0);
  bnprep_kernel<<<1, 256, 0, stream>>>(bn_gamma, bn_beta, bn_mean, bn_var, bnss);

  // 1) GRU scan -> hs (bf16)
  scan_kernel<0><<<B_ / 32, 256, 0, stream>>>(
      inputs, hsb, pGRU, gru_b, gru_b + 384, (const float*)nullptr,
      (const float*)nullptr, (float*)nullptr);

  // 2) attention: a1 = relu(emb @ W1 + b1)  (emb built on the fly)
  gemm_kernel<512, 2><<<B_ * T_ / 64, 256, 0, stream>>>(
      (const void*)hsb, pA1, att_b1, (void*)a1, inputs, (const float*)nullptr, 128, 1, 0, 1);
  //    a2 = relu(a1 @ W2 + b2)
  gemm_kernel<128, 0><<<B_ * T_ / 64, 256, 0, stream>>>(
      (const void*)a1, pA2, att_b2, (void*)a2,
      (const float*)nullptr, (const float*)nullptr, 64, 1, 1, 0);
  //    ats = sigmoid(a2 @ W3 + b3)
  rowdot_sigmoid_kernel<<<(B_ * T_ + 255) / 256, 256, 0, stream>>>(
      a2, 64, att_W3, att_b3, ats, B_ * T_);

  // 3) AUGRU scan -> h_final
  scan_kernel<1><<<B_ / 32, 256, 0, stream>>>(
      inputs, hsb, pAU, au_bu, au_br, au_bc, ats, hfin);

  // 4) BN + dense MLP + final sigmoid
  gemm_kernel<256, 3><<<B_ / 64, 256, 0, stream>>>(
      (const void*)hfin, pD1, d_b1, (void*)x1, inputs, bnss, 256, 2, 0, 0);
  gemm_kernel<256, 0><<<B_ / 64, 256, 0, stream>>>(
      (const void*)x1, pD2, d_b2, (void*)x2,
      (const float*)nullptr, (const float*)nullptr, 128, 2, 0, 0);
  rowdot_sigmoid_kernel<<<(B_ + 255) / 256, 256, 0, stream>>>(
      x2, 128, f_W, f_b, (float*)d_out, B_);
}
